// Attention_49460843380927
// MI455X (gfx1250) — matrix operand order repaired and checked by reading, not compiled, not run on hardware
//
#include <hip/hip_runtime.h>
#include <hip/hip_bf16.h>

typedef _Float16 v16h __attribute__((ext_vector_type(16)));
typedef float    v8f  __attribute__((ext_vector_type(8)));

union H16 {
    v16h v;
    _Float16 h[16];
    uint4 q2[2];
};

__device__ __forceinline__ v8f zero8() {
    v8f z;
    #pragma unroll
    for (int i = 0; i < 8; ++i) z[i] = 0.0f;
    return z;
}

// Spatial permutation: output position p reads input position perm(p).
// mode 0 (hs11):  full 32x32 rotations rho^r
// mode 1 (hs111): quadrant-local 16x16 rotations rho^r
__device__ __forceinline__ int permute_pos(int mode, int r, int p) {
    int i = p >> 5, j = p & 31;
    if (r == 0) return p;
    if (mode == 0) {
        if (r == 1) return (j << 5) | (31 - i);
        if (r == 2) return ((31 - i) << 5) | (31 - j);
        return ((31 - j) << 5) | i;               // r == 3
    } else {
        int qi = i & 16, qj = j & 16, li = i & 15, lj = j & 15;
        int ni, nj;
        if (r == 1)      { ni = lj;       nj = 15 - li; }
        else if (r == 2) { ni = 15 - li;  nj = 15 - lj; }
        else             { ni = 15 - lj;  nj = li;      }
        return ((qi | ni) << 5) | (qj | nj);
    }
}

// ---------------------------------------------------------------------------
// Kernel 1: transpose x[b][c][p] (f32) -> xT[(b<<10)+p][c] (f16), tiled in LDS
// ---------------------------------------------------------------------------
__global__ __launch_bounds__(256) void xpose_kernel(const float* __restrict__ x,
                                                    _Float16* __restrict__ xT) {
    __shared__ _Float16 tile[32][33];
    const int b  = blockIdx.z;
    const int c0 = blockIdx.y * 32;
    const int p0 = blockIdx.x * 32;
    const int tid = threadIdx.x;
    const int tr = tid >> 5, tc = tid & 31;     // tr: 0..7, tc: 0..31
    #pragma unroll
    for (int cc = 0; cc < 4; ++cc) {
        int cl = tr + cc * 8;
        tile[cl][tc] = (_Float16)x[(((size_t)b * 768 + c0 + cl) << 10) + p0 + tc];
    }
    __syncthreads();
    #pragma unroll
    for (int pp = 0; pp < 4; ++pp) {
        int pl = tr + pp * 8;
        xT[(((size_t)b << 10) + p0 + pl) * 768 + c0 + tc] = tile[tc][pl];
    }
}

// ---------------------------------------------------------------------------
// Kernel 2: convert weights f32 -> f16 (flat copy; W[r][o][c] stays K-contig)
// ---------------------------------------------------------------------------
__global__ __launch_bounds__(256) void wconv_kernel(const float* __restrict__ W,
                                                    _Float16* __restrict__ Wh, int n) {
    int i = (blockIdx.x * 256 + threadIdx.x) * 4;
    if (i + 3 < n) {
        #pragma unroll
        for (int j = 0; j < 4; ++j) Wh[i + j] = (_Float16)W[i + j];
    }
}

// ---------------------------------------------------------------------------
// Kernel 3: fused permuted GEMM  sum_r relu(A[perm_r(m)] * W_r^T + b_r)
//   mode 0: r=0..3, rot perms,   f16 out (q = hs11)
//   mode 1: r=4..7, quad perms,  f16 out (k = hs111)
//   mode 2: r=8, identity,       f32 out to d_out [b][o][p]
// Tile: BM=128, BN=128, BK=32, double-buffered LDS (one barrier per K-step).
// 256 threads = 8 waves; each wave computes a 64x32 patch (4x2 WMMA tiles).
// ---------------------------------------------------------------------------
#define BM 128
#define BN 128
#define BK 32
#define AST 40   // LDS row stride (halves): 80B, 16B-aligned rows, bank-staggered
#define KROUNDS (768 / BK)

__global__ __launch_bounds__(256) void gemm_kernel(const _Float16* __restrict__ Ag,
                                                   const _Float16* __restrict__ Wg,
                                                   const float* __restrict__ bias,
                                                   _Float16* __restrict__ outh,
                                                   float* __restrict__ outf,
                                                   int mode) {
    __shared__ __align__(16) _Float16 Asl[2][BM * AST];
    __shared__ __align__(16) _Float16 Bsl[2][BN * AST];

    const int tid  = threadIdx.x;
    const int lane = tid & 31;
    const int wid  = tid >> 5;
    const int m0 = blockIdx.y * BM;
    const int n0 = blockIdx.x * BN;
    const int b     = m0 >> 10;
    const int pbase = m0 & 1023;

    const int wm = (wid >> 2) * 64;   // wave row offset within block tile
    const int wn = (wid & 3) * 32;    // wave col offset

    // loader mapping: each thread moves 16 halves of A and 16 halves of B
    const int l_row = tid >> 1;            // 0..127
    const int l_seg = (tid & 1) * 16;      // 0 or 16 halves
    const int lofs  = l_row * AST + l_seg; // LDS offset (halves)

    const int ln   = lane & 15;
    const int s8   = (lane >> 4) * 8;
    const int kb16 = (lane >> 4) * 16;

    const int nr = (mode == 2) ? 1 : 4;

    v8f sum[4][2];
    #pragma unroll
    for (int t = 0; t < 4; ++t)
        #pragma unroll
        for (int u = 0; u < 2; ++u) sum[t][u] = zero8();

    for (int rr = 0; rr < nr; ++rr) {
        const int r = (mode == 0) ? rr : (mode == 1 ? 4 + rr : 8);
        const _Float16* __restrict__ Wr = Wg + (size_t)r * 768 * 768;

        v8f acc[4][2];
        #pragma unroll
        for (int t = 0; t < 4; ++t)
            #pragma unroll
            for (int u = 0; u < 2; ++u) acc[t][u] = zero8();

        // Permuted source row for the A loader (constant across K)
        const int srcp = (mode == 2) ? (pbase + l_row)
                                     : permute_pos(mode, rr, pbase + l_row);
        const _Float16* __restrict__ Arow =
            Ag + (((size_t)(b << 10) + srcp) * 768 + l_seg);
        const _Float16* __restrict__ Brow =
            Wr + ((size_t)(n0 + l_row) * 768 + l_seg);

        // ---- preload round 0 ----
        {
            const uint4* ga = (const uint4*)(Arow);
            const uint4* gb = (const uint4*)(Brow);
            uint4 a0 = ga[0], a1 = ga[1];
            uint4 b0 = gb[0], b1 = gb[1];
            *(uint4*)&Asl[0][lofs]     = a0;
            *(uint4*)&Asl[0][lofs + 8] = a1;
            *(uint4*)&Bsl[0][lofs]     = b0;
            *(uint4*)&Bsl[0][lofs + 8] = b1;
        }
        __syncthreads();

        for (int rd = 0; rd < KROUNDS; ++rd) {
            const int cur = rd & 1;
            uint4 na0, na1, nb0, nb1;
            const bool more = (rd + 1 < KROUNDS);
            if (more) {
                const int ks = (rd + 1) * BK;
                const uint4* ga = (const uint4*)(Arow + ks);
                const uint4* gb = (const uint4*)(Brow + ks);
                na0 = ga[0]; na1 = ga[1];
                nb0 = gb[0]; nb1 = gb[1];
            }

            // fragments from the active buffer
            H16 a[4], bf[2];
            #pragma unroll
            for (int t = 0; t < 4; ++t) {
                int row = wm + t * 16 + ln;
                a[t].q2[0] = *(const uint4*)&Asl[cur][row * AST + s8];
                a[t].q2[1] = *(const uint4*)&Asl[cur][row * AST + 16 + s8];
            }
            #pragma unroll
            for (int u = 0; u < 2; ++u) {
                int col = wn + u * 16 + ln;
                bf[u].q2[0] = *(const uint4*)&Bsl[cur][col * AST + s8];
                bf[u].q2[1] = *(const uint4*)&Bsl[cur][col * AST + 16 + s8];
            }
            #pragma unroll
            for (int t = 0; t < 4; ++t)
                #pragma unroll
                for (int u = 0; u < 2; ++u)
                    acc[t][u] = __builtin_amdgcn_wmma_f32_16x16x32_f16(
                        false, a[t].v, false, bf[u].v, (short)0, acc[t][u], false, false);

            // fill the inactive buffer for the next round
            if (more) {
                *(uint4*)&Asl[cur ^ 1][lofs]     = na0;
                *(uint4*)&Asl[cur ^ 1][lofs + 8] = na1;
                *(uint4*)&Bsl[cur ^ 1][lofs]     = nb0;
                *(uint4*)&Bsl[cur ^ 1][lofs + 8] = nb1;
            }
            __syncthreads();
        }

        // bias + ReLU per conv, then accumulate
        #pragma unroll
        for (int u = 0; u < 2; ++u) {
            int o = n0 + wn + u * 16 + ln;
            float bv = bias[r * 768 + o];
            #pragma unroll
            for (int t = 0; t < 4; ++t)
                #pragma unroll
                for (int v = 0; v < 8; ++v)
                    sum[t][u][v] += fmaxf(acc[t][u][v] + bv, 0.0f);
        }
    }

    // Store: lane holds fixed o, 8 consecutive p values
    const int rowHalf = (lane >> 4) * 8;
    #pragma unroll
    for (int t = 0; t < 4; ++t) {
        int p = pbase + wm + t * 16 + rowHalf;
        #pragma unroll
        for (int u = 0; u < 2; ++u) {
            int o = n0 + wn + u * 16 + ln;
            size_t base = ((size_t)(b * 768 + o) << 10) + p;
            if (mode == 2) {
                float4 f0 = make_float4(sum[t][u][0], sum[t][u][1], sum[t][u][2], sum[t][u][3]);
                float4 f1 = make_float4(sum[t][u][4], sum[t][u][5], sum[t][u][6], sum[t][u][7]);
                *(float4*)&outf[base]     = f0;
                *(float4*)&outf[base + 4] = f1;
            } else {
                union { uint4 q; _Float16 h[8]; } pk;
                #pragma unroll
                for (int v = 0; v < 8; ++v) pk.h[v] = (_Float16)sum[t][u][v];
                *(uint4*)&outh[base] = pk.q;
            }
        }
    }
}

// ---------------------------------------------------------------------------
// Kernel 4: attention. One wave per (b,c): att = q*k, softmax rows, out=att*q.
// q/k are f16 [idx=(b*768+c)][1024]. Output written in [m=(b<<10)+p][c] f16
// layout (A-matrix layout for the final GEMM).
// ---------------------------------------------------------------------------
__global__ __launch_bounds__(256) void attn_kernel(const _Float16* __restrict__ qb,
                                                   const _Float16* __restrict__ kb,
                                                   _Float16* __restrict__ A2) {
    __shared__ __align__(16) _Float16 qs[8][32 * 40];
    __shared__ __align__(16) _Float16 ks[8][32 * 40];
    __shared__ __align__(16) _Float16 as[8][32 * 40];

    const int tid  = threadIdx.x;
    const int lane = tid & 31;
    const int wid  = tid >> 5;
    const int idx  = blockIdx.x * 8 + wid;      // b*768 + c
    const int b = idx / 768;
    const int c = idx - b * 768;

    const _Float16* __restrict__ qp = qb + (size_t)idx * 1024;
    const _Float16* __restrict__ kp = kb + (size_t)idx * 1024;
    _Float16* qsl = qs[wid];
    _Float16* ksl = ks[wid];
    _Float16* asl = as[wid];

    const int ln   = lane & 15;
    const int s8   = (lane >> 4) * 8;
    const int kb16 = (lane >> 4) * 16;

    // stage q,k into LDS: lane = spatial row i, 32 halves each
    {
        const uint4* q4 = (const uint4*)(qp + lane * 32);
        const uint4* k4 = (const uint4*)(kp + lane * 32);
        #pragma unroll
        for (int t = 0; t < 4; ++t) {
            *(uint4*)&qsl[lane * 40 + t * 8] = q4[t];
            *(uint4*)&ksl[lane * 40 + t * 8] = k4[t];
        }
    }
    __syncthreads();

    // att = q (i x k) * k (k x j)
    H16 aq[2], bk[2];
    #pragma unroll
    for (int it = 0; it < 2; ++it) {
        int row = it * 16 + ln;
        aq[it].q2[0] = *(const uint4*)&qsl[row * 40 + s8];
        aq[it].q2[1] = *(const uint4*)&qsl[row * 40 + 16 + s8];
    }
    #pragma unroll
    for (int jt = 0; jt < 2; ++jt) {
        int col = jt * 16 + ln;
        #pragma unroll
        for (int e = 0; e < 16; ++e) bk[jt].h[e] = ksl[(s8 + (e & 7) + ((e >> 3) << 4)) * 40 + col];
    }
    v8f att[2][2];
    #pragma unroll
    for (int it = 0; it < 2; ++it)
        #pragma unroll
        for (int jt = 0; jt < 2; ++jt) {
            att[it][jt] = zero8();
            att[it][jt] = __builtin_amdgcn_wmma_f32_16x16x32_f16(
                false, aq[it].v, false, bk[jt].v, (short)0, att[it][jt], false, false);
        }

    // softmax per row (row spread over 16 lanes x 2 jt tiles), write f16 -> asl
    #pragma unroll
    for (int it = 0; it < 2; ++it) {
        int rbase = it * 16 + (lane >> 4) * 8;
        #pragma unroll
        for (int v = 0; v < 8; ++v) {
            float x0 = att[it][0][v], x1 = att[it][1][v];
            float m = fmaxf(x0, x1);
            #pragma unroll
            for (int msk = 1; msk <= 8; msk <<= 1) m = fmaxf(m, __shfl_xor(m, msk, 32));
            float e0 = __expf(x0 - m), e1 = __expf(x1 - m);
            float s = e0 + e1;
            #pragma unroll
            for (int msk = 1; msk <= 8; msk <<= 1) s += __shfl_xor(s, msk, 32);
            float inv = 1.0f / s;
            asl[(rbase + v) * 40 + ln]      = (_Float16)(e0 * inv);
            asl[(rbase + v) * 40 + 16 + ln] = (_Float16)(e1 * inv);
        }
    }
    __syncthreads();

    // out = attn (i x j) * v (j x h), v = q
    H16 aa[2], bv[2];
    #pragma unroll
    for (int it = 0; it < 2; ++it) {
        int row = it * 16 + ln;
        aa[it].q2[0] = *(const uint4*)&asl[row * 40 + s8];
        aa[it].q2[1] = *(const uint4*)&asl[row * 40 + 16 + s8];
    }
    #pragma unroll
    for (int ht = 0; ht < 2; ++ht) {
        int col = ht * 16 + ln;
        #pragma unroll
        for (int e = 0; e < 16; ++e) bv[ht].h[e] = qsl[(s8 + (e & 7) + ((e >> 3) << 4)) * 40 + col];
    }
    v8f o[2][2];
    #pragma unroll
    for (int it = 0; it < 2; ++it)
        #pragma unroll
        for (int ht = 0; ht < 2; ++ht) {
            o[it][ht] = zero8();
            o[it][ht] = __builtin_amdgcn_wmma_f32_16x16x32_f16(
                false, aa[it].v, false, bv[ht].v, (short)0, o[it][ht], false, false);
        }

    // scatter to A2[(b<<10)+p][c], p = i*32 + h
    const int pb = b << 10;
    #pragma unroll
    for (int it = 0; it < 2; ++it) {
        int rbase = it * 16 + (lane >> 4) * 8;
        #pragma unroll
        for (int ht = 0; ht < 2; ++ht) {
            int h = ht * 16 + ln;
            #pragma unroll
            for (int v = 0; v < 8; ++v) {
                int p = (rbase + v) * 32 + h;
                A2[(size_t)(pb + p) * 768 + c] = (_Float16)o[it][ht][v];
            }
        }
    }
}

// ---------------------------------------------------------------------------
extern "C" void kernel_launch(void* const* d_in, const int* in_sizes, int n_in,
                              void* d_out, int out_size, void* d_ws, size_t ws_size,
                              hipStream_t stream) {
    (void)in_sizes; (void)n_in; (void)out_size; (void)ws_size;
    const float* x  = (const float*)d_in[0];
    const float* Ws = (const float*)d_in[1];
    const float* bs = (const float*)d_in[2];
    float* out = (float*)d_out;

    // workspace: xT (48MB f16) | Wh (10MB f16) | A2 (48MB f16)
    _Float16* xT = (_Float16*)d_ws;
    _Float16* Wh = xT + (size_t)32768 * 768;
    _Float16* A2 = Wh + (size_t)9 * 768 * 768;
    // q/k live inside d_out (2 x 48MB f16 == 100MB, exactly the f32 out size);
    // they are dead before the final GEMM overwrites d_out.
    _Float16* qb = (_Float16*)d_out;
    _Float16* kbuf = qb + (size_t)24576 * 1024;

    xpose_kernel<<<dim3(32, 24, 32), 256, 0, stream>>>(x, xT);
    wconv_kernel<<<dim3((9 * 768 * 768) / (256 * 4)), 256, 0, stream>>>(Ws, Wh, 9 * 768 * 768);
    gemm_kernel<<<dim3(768 / BN, 32768 / BM), 256, 0, stream>>>(xT, Wh, bs, qb, nullptr, 0);
    gemm_kernel<<<dim3(768 / BN, 32768 / BM), 256, 0, stream>>>(xT, Wh, bs, kbuf, nullptr, 1);
    attn_kernel<<<dim3(24576 / 8), 256, 0, stream>>>(qb, kbuf, A2);
    gemm_kernel<<<dim3(768 / BN, 32768 / BM), 256, 0, stream>>>(A2, Wh, bs, nullptr, out, 2);
}
